// SelfAttention1D_86242943304186
// MI455X (gfx1250) — matrix operand order repaired and checked by reading, not compiled, not run on hardware
//
#include <hip/hip_runtime.h>
#include <hip/hip_bf16.h>

#define B_ 16
#define C_ 256
#define W_ 2048
#define D_ 32   // C/8

typedef __attribute__((ext_vector_type(16))) __bf16 v16bf;
typedef __attribute__((ext_vector_type(8)))  float  v8f;

union BFrag {
    unsigned short u16[16];
    uint4          u128[2];
    v16bf          v;
};

__device__ __forceinline__ unsigned short f2bf(float f) {
    __bf16 h = (__bf16)f;                 // native cvt on gfx1250
    unsigned short u; __builtin_memcpy(&u, &h, 2);
    return u;
}

__device__ __forceinline__ v8f wmma_bf16(const BFrag& a, const BFrag& b, v8f c) {
    return __builtin_amdgcn_wmma_f32_16x16x32_bf16(
        false, a.v, false, b.v, (short)0, c, false, false);
}

// ---------------------------------------------------------------------------
// Prep kernel A: fp32 -> bf16 copy (weights).
// ---------------------------------------------------------------------------
__global__ __launch_bounds__(256) void convw_kernel(
    const float* __restrict__ src, unsigned short* __restrict__ dst, int n)
{
    int i = blockIdx.x * 256 + threadIdx.x;
    if (i < n) dst[i] = f2bf(src[i]);
}

// ---------------------------------------------------------------------------
// Prep kernel B: x [b][c][w] fp32 -> xT [b][w][c] bf16 (LDS tile transpose).
// ---------------------------------------------------------------------------
__global__ __launch_bounds__(256) void transx_kernel(
    const float* __restrict__ x, unsigned short* __restrict__ xT)
{
    __shared__ unsigned short tile[32][33];
    const int b  = blockIdx.z;
    const int c0 = blockIdx.y * 32;
    const int w0 = blockIdx.x * 32;
    const int t0 = threadIdx.x & 31;
    const int t1 = threadIdx.x >> 5;      // 0..7

    const float* xb = x + ((size_t)b * C_ + c0) * W_;
    #pragma unroll
    for (int cc = 0; cc < 32; cc += 8)
        tile[cc + t1][t0] = f2bf(xb[(size_t)(cc + t1) * W_ + w0 + t0]);
    __syncthreads();
    unsigned short* xtb = xT + ((size_t)b * W_ + w0) * C_;
    #pragma unroll
    for (int ww = 0; ww < 32; ww += 8)
        xtb[(size_t)(ww + t1) * C_ + c0 + t0] = tile[t0][ww + t1];
}

// ---------------------------------------------------------------------------
// Kernel 1: QKV projection, all-bf16 fragment loads (aligned b128 only).
// 20 row-tiles (Q:2, K:2, V:16); 10 waves x 2 tiles; 32-wide w strip.
// Q,K stored transposed bf16 [b][w][32]; V stored natural bf16 [b][c][w].
// ---------------------------------------------------------------------------
__global__ __launch_bounds__(320) void proj_kernel(
    const unsigned short* __restrict__ xT,
    const unsigned short* __restrict__ wqb, const float* __restrict__ bq,
    const unsigned short* __restrict__ wkb, const float* __restrict__ bk,
    const unsigned short* __restrict__ wvb, const float* __restrict__ bv,
    unsigned short* __restrict__ Qt, unsigned short* __restrict__ Kt,
    unsigned short* __restrict__ Vw)
{
    const int lane = threadIdx.x & 31;
    const int wid  = threadIdx.x >> 5;           // 0..9
    const int lo   = lane & 15;
    const int hi   = lane >> 4;
    const int b    = blockIdx.y;
    const int w0   = blockIdx.x * 32;
    const int kb    = hi * 8;                    // unused (A-frag also uses kbase)
    const int kbase = hi * 16;                   // B-frag K base

    const unsigned short* xr0 = xT + ((size_t)b * W_ + w0 + lo) * C_;
    const unsigned short* xr1 = xr0 + (size_t)16 * C_;

    for (int tt = 0; tt < 2; ++tt) {
        const int T = wid * 2 + tt;              // 0..19
        const unsigned short* wmat; const float* bias; int o0, kind;
        if (T < 2)      { wmat = wqb; bias = bq; o0 = T * 16;       kind = 0; }
        else if (T < 4) { wmat = wkb; bias = bk; o0 = (T - 2) * 16; kind = 1; }
        else            { wmat = wvb; bias = bv; o0 = (T - 4) * 16; kind = 2; }

        v8f acc0 = {}; v8f acc1 = {};
        const unsigned short* wrow = wmat + (size_t)(o0 + lo) * C_;

        #pragma unroll
        for (int k0 = 0; k0 < C_; k0 += 32) {
            BFrag af, bx0, bx1;
            af.u128[0]  = *(const uint4*)(wrow + k0 + kbase);
            af.u128[1]  = *(const uint4*)(wrow + k0 + kbase + 8);
            bx0.u128[0] = *(const uint4*)(xr0 + k0 + kbase);
            bx0.u128[1] = *(const uint4*)(xr0 + k0 + kbase + 8);
            bx1.u128[0] = *(const uint4*)(xr1 + k0 + kbase);
            bx1.u128[1] = *(const uint4*)(xr1 + k0 + kbase + 8);
            acc0 = wmma_bf16(af, bx0, acc0);
            acc1 = wmma_bf16(af, bx1, acc1);
        }

        // Bias: D rows o = o0 + 8*hi + r
        {
            float4 b0 = *(const float4*)(bias + o0 + 8 * hi);
            float4 b1 = *(const float4*)(bias + o0 + 8 * hi + 4);
            float bb[8] = { b0.x, b0.y, b0.z, b0.w, b1.x, b1.y, b1.z, b1.w };
            #pragma unroll
            for (int e = 0; e < 8; ++e) { acc0[e] += bb[e]; acc1[e] += bb[e]; }
        }

        if (kind < 2) {
            unsigned short* dst = (kind == 0) ? Qt : Kt;
            #pragma unroll
            for (int s = 0; s < 2; ++s) {
                v8f a = s ? acc1 : acc0;
                unsigned short pk[8];
                #pragma unroll
                for (int e = 0; e < 8; ++e) pk[e] = f2bf(a[e]);
                const int w = w0 + s * 16 + lo;
                uint4 val; __builtin_memcpy(&val, pk, 16);
                *(uint4*)(dst + ((size_t)b * W_ + w) * D_ + o0 + 8 * hi) = val;
            }
        } else {
            #pragma unroll
            for (int s = 0; s < 2; ++s) {
                v8f a = s ? acc1 : acc0;
                const int w = w0 + s * 16 + lo;
                #pragma unroll
                for (int e = 0; e < 8; ++e) {
                    const int c = o0 + e + 8 * hi;
                    Vw[((size_t)b * C_ + c) * W_ + w] = f2bf(a[e]);
                }
            }
        }
    }
}

// ---------------------------------------------------------------------------
// Kernel 2: fused flash attention + residual.
// Grid (W/32, B); 4 waves; each wave: 64 channels (4 c-tiles) x 32 queries
// (2 i-subtiles).  Per j-step per wave: 4 S-WMMA (redundant across waves,
// avoids workgroup barriers) + 8 P.V WMMA.  P transposed through per-wave
// private LDS (in-order DS + s_wait_dscnt only, no s_barrier).
// ---------------------------------------------------------------------------
__global__ __launch_bounds__(128) void attn_kernel(
    const unsigned short* __restrict__ Qt, const unsigned short* __restrict__ Kt,
    const unsigned short* __restrict__ Vw, const float* __restrict__ x,
    const float* __restrict__ gamma, float* __restrict__ out)
{
    __shared__ __align__(16) unsigned short Plds[4][2][16 * 32];
    __shared__ __align__(16) float          Fbc[4][2][16];

    const int lane = threadIdx.x & 31;
    const int wid  = threadIdx.x >> 5;    // 0..3
    const int lo   = lane & 15;
    const int hi   = lane >> 4;
    const int b    = blockIdx.y;
    const int i0   = blockIdx.x * 32;     // 32-query tile
    const int c0   = wid * 64;            // this wave's 64 channels
    const int kbA   = hi * 8;
    const int kbase = hi * 16;

    // Q A-fragments for both i-subtiles
    BFrag qf[2];
    #pragma unroll
    for (int u = 0; u < 2; ++u) {
        const unsigned short* qrow = Qt + ((size_t)b * W_ + i0 + u * 16 + lo) * D_;
        qf[u].u128[0] = *(const uint4*)(qrow + kbase);
        qf[u].u128[1] = *(const uint4*)(qrow + kbase + 8);
    }

    float m[2][8], l[2][8];
    #pragma unroll
    for (int u = 0; u < 2; ++u)
        #pragma unroll
        for (int r = 0; r < 8; ++r) { m[u][r] = -3.0e38f; l[u][r] = 0.0f; }
    v8f acc[4][2];
    #pragma unroll
    for (int t = 0; t < 4; ++t)
        #pragma unroll
        for (int u = 0; u < 2; ++u) acc[t][u] = (v8f){};

    const unsigned short* krow  = Kt + ((size_t)b * W_ + lo) * D_ + kbase;
    const unsigned short* vbase = Vw + ((size_t)b * C_ + c0 + lo) * W_;

    for (int j0 = 0; j0 < W_; j0 += 32) {
        // K B-fragments, columns j0+lo and j0+16+lo
        BFrag kf0, kf1;
        const unsigned short* kr0 = krow + (size_t)j0 * D_;
        const unsigned short* kr1 = kr0 + 16 * D_;
        kf0.u128[0] = *(const uint4*)(kr0);
        kf0.u128[1] = *(const uint4*)(kr0 + 8);
        kf1.u128[0] = *(const uint4*)(kr1);
        kf1.u128[1] = *(const uint4*)(kr1 + 8);

        if (j0 + 32 < W_) {
            __builtin_prefetch(kr0 + 32 * D_, 0, 1);
            #pragma unroll
            for (int t = 0; t < 4; ++t)
                __builtin_prefetch(vbase + (size_t)(t * 16) * W_ + j0 + 32, 0, 1);
        }

        v8f z = {};
        v8f s[2][2];
        #pragma unroll
        for (int u = 0; u < 2; ++u) {
            s[u][0] = wmma_bf16(qf[u], kf0, z);
            s[u][1] = wmma_bf16(qf[u], kf1, z);
        }

        // Online softmax per i-subtile; rows i = r + 8*hi live on 16 lanes.
        float fr[2][8];
        #pragma unroll
        for (int u = 0; u < 2; ++u) {
            #pragma unroll
            for (int r = 0; r < 8; ++r) {
                float a0 = s[u][0][r], a1 = s[u][1][r];
                float mx = fmaxf(a0, a1);
                mx = fmaxf(mx, __shfl_xor(mx, 1, 32));
                mx = fmaxf(mx, __shfl_xor(mx, 2, 32));
                mx = fmaxf(mx, __shfl_xor(mx, 4, 32));
                mx = fmaxf(mx, __shfl_xor(mx, 8, 32));
                float nm = fmaxf(m[u][r], mx);
                float p0 = __expf(a0 - nm);
                float p1 = __expf(a1 - nm);
                float rs = p0 + p1;
                rs += __shfl_xor(rs, 1, 32);
                rs += __shfl_xor(rs, 2, 32);
                rs += __shfl_xor(rs, 4, 32);
                rs += __shfl_xor(rs, 8, 32);
                float f = __expf(m[u][r] - nm);
                l[u][r] = l[u][r] * f + rs;
                m[u][r] = nm;
                fr[u][r] = f;
                const int il = r + 8 * hi;
                Plds[wid][u][il * 32 + lo]      = f2bf(p0);
                Plds[wid][u][il * 32 + 16 + lo] = f2bf(p1);
            }
        }
        if (lo == 0) {
            #pragma unroll
            for (int u = 0; u < 2; ++u)
                #pragma unroll
                for (int r = 0; r < 8; ++r) Fbc[wid][u][8 * hi + r] = fr[u][r];
        }
        asm volatile("s_wait_dscnt 0" ::: "memory");   // wave-private LDS fence

        float fc[2] = { Fbc[wid][0][lo], Fbc[wid][1][lo] };

        BFrag pf[2];
        #pragma unroll
        for (int u = 0; u < 2; ++u) {
            const unsigned short* pr = &Plds[wid][u][lo * 32 + kbase];
            pf[u].u128[0] = *(const uint4*)(pr);
            pf[u].u128[1] = *(const uint4*)(pr + 8);
        }

        #pragma unroll
        for (int t = 0; t < 4; ++t) {
            BFrag vf;
            const unsigned short* vr = vbase + (size_t)(t * 16) * W_ + j0;
            vf.u128[0] = *(const uint4*)(vr + kbase);
            vf.u128[1] = *(const uint4*)(vr + kbase + 8);
            #pragma unroll
            for (int u = 0; u < 2; ++u) {
                #pragma unroll
                for (int e = 0; e < 8; ++e) acc[t][u][e] *= fc[u];
                acc[t][u] = wmma_bf16(vf, pf[u], acc[t][u]);
            }
        }
    }

    // Finalize: broadcast 1/l per query column, gamma + residual, store fp32.
    asm volatile("s_wait_dscnt 0" ::: "memory");
    if (lo == 0) {
        #pragma unroll
        for (int u = 0; u < 2; ++u)
            #pragma unroll
            for (int r = 0; r < 8; ++r) Fbc[wid][u][8 * hi + r] = l[u][r];
    }
    asm volatile("s_wait_dscnt 0" ::: "memory");
    const float linv[2] = { 1.0f / Fbc[wid][0][lo], 1.0f / Fbc[wid][1][lo] };
    const float g = gamma[0];
    #pragma unroll
    for (int t = 0; t < 4; ++t) {
        const int cbase = c0 + t * 16 + 8 * hi;
        #pragma unroll
        for (int u = 0; u < 2; ++u) {
            const int w = i0 + u * 16 + lo;
            #pragma unroll
            for (int r = 0; r < 8; ++r) {
                const size_t idx = ((size_t)b * C_ + cbase + r) * W_ + w;
                out[idx] = g * (acc[t][u][r] * linv[u]) + x[idx];
            }
        }
    }
}

// ---------------------------------------------------------------------------
extern "C" void kernel_launch(void* const* d_in, const int* in_sizes, int n_in,
                              void* d_out, int out_size, void* d_ws, size_t ws_size,
                              hipStream_t stream) {
    const float* x     = (const float*)d_in[0];
    const float* wq    = (const float*)d_in[1];
    const float* bq    = (const float*)d_in[2];
    const float* wk    = (const float*)d_in[3];
    const float* bk    = (const float*)d_in[4];
    const float* wv    = (const float*)d_in[5];
    const float* bv    = (const float*)d_in[6];
    const float* gamma = (const float*)d_in[7];
    float* out = (float*)d_out;

    // Workspace (bf16 ushorts): Qt | Kt | V | xT | wqb | wkb | wvb  (~38 MB)
    unsigned short* Qt  = (unsigned short*)d_ws;
    unsigned short* Kt  = Qt  + (size_t)B_ * W_ * D_;
    unsigned short* Vw  = Kt  + (size_t)B_ * W_ * D_;
    unsigned short* xT  = Vw  + (size_t)B_ * C_ * W_;
    unsigned short* wqb = xT  + (size_t)B_ * C_ * W_;
    unsigned short* wkb = wqb + (size_t)D_ * C_;
    unsigned short* wvb = wkb + (size_t)D_ * C_;

    convw_kernel<<<dim3((D_ * C_ + 255) / 256), 256, 0, stream>>>(wq, wqb, D_ * C_);
    convw_kernel<<<dim3((D_ * C_ + 255) / 256), 256, 0, stream>>>(wk, wkb, D_ * C_);
    convw_kernel<<<dim3((C_ * C_ + 255) / 256), 256, 0, stream>>>(wv, wvb, C_ * C_);
    transx_kernel<<<dim3(W_ / 32, C_ / 32, B_), 256, 0, stream>>>(x, xT);

    proj_kernel<<<dim3(W_ / 32, B_), 320, 0, stream>>>(
        xT, wqb, bq, wkb, bk, wvb, bv, Qt, Kt, Vw);

    attn_kernel<<<dim3(W_ / 32, B_), 128, 0, stream>>>(Qt, Kt, Vw, x, gamma, out);
}
